// RatingLayer_6846177870362
// MI455X (gfx1250) — matrix operand order repaired and checked by reading, not compiled, not run on hardware
//
#include <hip/hip_runtime.h>
#include <stdint.h>

#define NI 2048
#define H  4096
#define BS 4096
#define G6 6

#define TM 64
#define TN 32
#define TK 32
#define LDA 40                      // padded A row (elems)
#define LDB 40                      // padded B row (elems)
#define A_ELEMS (TM * LDA)          // 2560 bf16
#define B_ELEMS (G6 * TN * LDB)     // 7680 bf16
#define BUF_ELEMS (A_ELEMS + B_ELEMS)

#define AS1 __attribute__((address_space(1)))
#define AS3 __attribute__((address_space(3)))

// Pointee type required by the async-DMA builtin (from hipcc diagnostic).
typedef int v4i_vs __attribute__((__vector_size__(16)));

// CDNA5 async global->LDS DMA path (ASYNCcnt), with guaranteed fallback.
#if defined(__gfx1250__) && \
    __has_builtin(__builtin_amdgcn_global_load_async_to_lds_b128) && \
    __has_builtin(__builtin_amdgcn_s_wait_asynccnt)
#define USE_ASYNC 1
#define NBUF 3
#else
#define USE_ASYNC 0
#define NBUF 2
#endif

typedef __attribute__((ext_vector_type(16))) __bf16 v16bf;
typedef __attribute__((ext_vector_type(8)))  float  v8f;

__device__ __forceinline__ uint16_t f2bf(float f) {
  uint32_t u = __float_as_uint(f);
  uint32_t r = (u + 0x7FFFu + ((u >> 16) & 1u)) >> 16;  // RNE
  return (uint16_t)r;
}

// Pack W_ih (half-swap of columns folded in: k -> k^2048) and W_hh into one
// bf16 matrix Wb[6][H][H]; gates: 0=i_r 1=i_z 2=i_n 3=h_r 4=h_z 5=h_n.
__global__ void k_pack_weights(const float* __restrict__ Wih,
                               const float* __restrict__ Whh,
                               uint16_t* __restrict__ Wb) {
  size_t i4 = (size_t)blockIdx.x * blockDim.x + threadIdx.x;
  size_t e = i4 * 4;
  const size_t HH = (size_t)H * H;
  if (e >= (size_t)G6 * HH) return;
  size_t g = e / HH;
  size_t rem = e - g * HH;
  size_t j = rem / H;
  size_t k = rem - j * H;
  const float* src;
  if (g < 3) src = Wih + ((size_t)g * H + j) * H + (k ^ (size_t)NI);
  else       src = Whh + ((size_t)(g - 3) * H + j) * H + k;
  float4 v = *(const float4*)src;
  ushort4 o;
  o.x = f2bf(v.x); o.y = f2bf(v.y); o.z = f2bf(v.z); o.w = f2bf(v.w);
  *(ushort4*)(Wb + e) = o;
}

__global__ void k_f32_to_bf16(const float* __restrict__ x,
                              uint16_t* __restrict__ y, size_t n4) {
  size_t i = (size_t)blockIdx.x * blockDim.x + threadIdx.x;
  if (i >= n4) return;
  float4 v = ((const float4*)x)[i];
  ushort4 o;
  o.x = f2bf(v.x); o.y = f2bf(v.y); o.z = f2bf(v.z); o.w = f2bf(v.w);
  ((ushort4*)y)[i] = o;
}

// One GRU round, fully fused: 6-gate GEMM (bf16 WMMA, f32 acc) + gate math.
__launch_bounds__(256)
__global__ void k_gru_round(const uint16_t* __restrict__ Abf,
                            const uint16_t* __restrict__ Wb,
                            const float* __restrict__ hprev,
                            const float* __restrict__ b_ih,
                            const float* __restrict__ b_hh,
                            float* __restrict__ hnext,
                            uint16_t* __restrict__ hnextb) {
  __shared__ __align__(16) uint16_t lds[NBUF][BUF_ELEMS];

  const int tid  = threadIdx.x;
  const int lane = tid & 31;
  const int wv   = tid >> 5;              // 8 waves: 4 (M) x 2 (N)
  const int m0   = blockIdx.x * TM;       // M is the FAST grid dim (L2 reuse of Wb)
  const int j0   = blockIdx.y * TN;

  // ---- cooperative load plan: A tile 64x32 (1 chunk/thread),
  //      B tile 6x32x32 n-major (3 chunks/thread), chunks are 16B ----
  const int arow = tid >> 2, aq = tid & 3;
  uint16_t* gAe = (uint16_t*)(Abf + (size_t)(m0 + arow) * H + aq * 8);
  const int aOff = arow * LDA + aq * 8;

  uint16_t* gBe[3];
  int bOff[3];
#pragma unroll
  for (int i = 0; i < 3; ++i) {
    int c = tid + 256 * i;                // 768 chunks
    int rowb = c >> 2;                    // g*32+n
    int q = c & 3;
    int g = rowb >> 5;
    int n = rowb & 31;
    gBe[i] = (uint16_t*)(Wb + ((size_t)g * H + j0 + n) * (size_t)H + q * 8);
    bOff[i] = A_ELEMS + (g * TN + n) * LDB + q * 8;
  }

  // ---- accumulators: 6 gates x one 16x16 f32 tile per wave ----
  const v8f vzero = {0.f, 0.f, 0.f, 0.f, 0.f, 0.f, 0.f, 0.f};
  v8f acc[G6];
#pragma unroll
  for (int g = 0; g < G6; ++g) acc[g] = vzero;

  // fragment lane addressing (ISA 16-bit A 16x32 / B 32x16 layouts)
  const int frow = (wv & 3) * 16 + (lane & 15);      // A row in LDS tile
  const int nrow = (wv >> 2) * 16 + (lane & 15);     // B n-row in LDS tile
  const int fka  = (lane >> 4) * 8;                  // A K base (lo half; +16 hi)
  const int fkb  = (lane >> 4) * 8;                  // B K base (lo half; +16 hi)

  auto compute = [&](int buf) {
    union Frag { uint4 u[2]; v16bf v; };
    const uint16_t* Lb = lds[buf];
    Frag af;
    af.u[0] = *(const uint4*)&Lb[frow * LDA + fka];
    af.u[1] = *(const uint4*)&Lb[frow * LDA + fka + 16];
    Frag bq[G6];
#pragma unroll
    for (int g = 0; g < G6; ++g) {                   // preload all 6 B frags
      const int off = A_ELEMS + (g * TN + nrow) * LDB + fkb;
      bq[g].u[0] = *(const uint4*)&Lb[off];
      bq[g].u[1] = *(const uint4*)&Lb[off + 16];
    }
#pragma unroll
    for (int g = 0; g < G6; ++g)
      acc[g] = __builtin_amdgcn_wmma_f32_16x16x32_bf16(
          false, af.v, false, bq[g].v, (short)0, acc[g], false, false);
  };

  const int KT = H / TK;                             // 128 k-steps

#if USE_ASYNC
  // 3-stage async DMA pipeline: tile t computes while t+1 is landing and t+2
  // is being issued. Each tile = 4 async-load instructions per wave, so
  // s_wait_asynccnt(4) == "everything except the newest tile has landed".
  auto issue = [&](int kt, int buf) {
    __builtin_amdgcn_global_load_async_to_lds_b128(
        (AS1 v4i_vs*)(gAe + (size_t)kt * TK),
        (AS3 v4i_vs*)&lds[buf][aOff], 0, 0);
#pragma unroll
    for (int i = 0; i < 3; ++i)
      __builtin_amdgcn_global_load_async_to_lds_b128(
          (AS1 v4i_vs*)(gBe[i] + (size_t)kt * TK),
          (AS3 v4i_vs*)&lds[buf][bOff[i]], 0, 0);
  };

  issue(0, 0);
  issue(1, 1);
  // steady state: branch-free body, unrolled x3 so buffer ids are constants
  for (int kt = 0; kt < KT - 2; kt += 3) {           // kt = 0,3,...,123
    __builtin_amdgcn_s_wait_asynccnt(4); __syncthreads();
    issue(kt + 2, 2); compute(0);
    __builtin_amdgcn_s_wait_asynccnt(4); __syncthreads();
    issue(kt + 3, 0); compute(1);
    __builtin_amdgcn_s_wait_asynccnt(4); __syncthreads();
    issue(kt + 4, 1); compute(2);
  }
  // drain: tiles KT-2 (buf 0) and KT-1 (buf 1)
  __builtin_amdgcn_s_wait_asynccnt(4); __syncthreads();
  compute(0);
  __builtin_amdgcn_s_wait_asynccnt(0); __syncthreads();
  compute(1);
#else
  uint4 ra, rb[3];
  auto fetch = [&](int kt) {
    ra = *(const uint4*)(gAe + (size_t)kt * TK);
#pragma unroll
    for (int i = 0; i < 3; ++i) rb[i] = *(const uint4*)(gBe[i] + (size_t)kt * TK);
  };
  auto stage = [&](int buf) {
    *(uint4*)&lds[buf][aOff] = ra;
#pragma unroll
    for (int i = 0; i < 3; ++i) *(uint4*)&lds[buf][bOff[i]] = rb[i];
  };
  fetch(0); stage(0); __syncthreads();
  for (int kt = 0; kt < KT - 1; ++kt) {
    fetch(kt + 1);
    compute(kt & 1);
    stage((kt + 1) & 1);
    __syncthreads();
  }
  compute((KT - 1) & 1);
#endif

  // ---- fused GRU epilogue ----
  const int mrow = m0 + (wv & 3) * 16 + (lane >> 4) * 8;
  const int nn   = j0 + (wv >> 2) * 16 + (lane & 15);
  const float bir = b_ih[nn], biz = b_ih[H + nn], bin = b_ih[2 * H + nn];
  const float bhr = b_hh[nn], bhz = b_hh[H + nn], bhn = b_hh[2 * H + nn];
#pragma unroll
  for (int r = 0; r < 8; ++r) {
    const int mm = mrow + r;
    float ir  = acc[0][r] + bir;
    float iz  = acc[1][r] + biz;
    float inn = acc[2][r] + bin;
    float hr  = acc[3][r] + bhr;
    float hz  = acc[4][r] + bhz;
    float hn  = acc[5][r] + bhn;
    float rg = 1.0f / (1.0f + __expf(-(ir + hr)));
    float zg = 1.0f / (1.0f + __expf(-(iz + hz)));
    float ng = tanhf(inn + rg * hn);
    float hp = hprev[(size_t)mm * H + nn];
    float hv = (1.0f - zg) * ng + zg * hp;
    hnext[(size_t)mm * H + nn]  = hv;
    hnextb[(size_t)mm * H + nn] = f2bf(hv);
  }
}

// out[b] = dot(h2[b,:], fc_w) + fc_b  (one wave per row)
__global__ void k_fc(const float* __restrict__ h, const float* __restrict__ w,
                     const float* __restrict__ b, float* __restrict__ out) {
  const int wv = threadIdx.x >> 5, lane = threadIdx.x & 31;
  const int row = blockIdx.x * 8 + wv;
  const float4* hp = (const float4*)(h + (size_t)row * H);
  const float4* wp = (const float4*)w;
  float s = 0.f;
  for (int i = lane; i < H / 4; i += 32) {
    float4 a = hp[i], c = wp[i];
    s += a.x * c.x + a.y * c.y + a.z * c.z + a.w * c.w;
  }
#pragma unroll
  for (int o = 16; o > 0; o >>= 1) s += __shfl_xor(s, o, 32);
  if (lane == 0) out[row] = s + b[0];
}

extern "C" void kernel_launch(void* const* d_in, const int* in_sizes, int n_in,
                              void* d_out, int out_size, void* d_ws, size_t ws_size,
                              hipStream_t stream) {
  const float* features = (const float*)d_in[0];   // [BS,2,NI] == [BS,H]
  const float* W_ih = (const float*)d_in[1];
  const float* b_ih = (const float*)d_in[2];
  const float* W_hh = (const float*)d_in[3];
  const float* b_hh = (const float*)d_in[4];
  const float* fc_w = (const float*)d_in[5];
  const float* fc_b = (const float*)d_in[6];
  float* out = (float*)d_out;

  char* ws = (char*)d_ws;
  uint16_t* Wb = (uint16_t*)ws;                                 // 6*H*H bf16
  size_t off = (size_t)G6 * H * H * sizeof(uint16_t);
  uint16_t* hbf0 = (uint16_t*)(ws + off); off += (size_t)BS * H * 2;
  uint16_t* hbf1 = (uint16_t*)(ws + off); off += (size_t)BS * H * 2;
  float* h1 = (float*)(ws + off); off += (size_t)BS * H * 4;
  float* h2 = (float*)(ws + off);

  {
    size_t n4 = (size_t)G6 * H * H / 4;
    k_pack_weights<<<dim3((unsigned)((n4 + 255) / 256)), dim3(256), 0, stream>>>(
        W_ih, W_hh, Wb);
  }
  {
    size_t n4 = (size_t)BS * H / 4;
    k_f32_to_bf16<<<dim3((unsigned)((n4 + 255) / 256)), dim3(256), 0, stream>>>(
        features, hbf0, n4);
  }
  dim3 grid(BS / TM, H / TN);   // (64, 128): M fastest -> weight slice L2 reuse
  k_gru_round<<<grid, dim3(256), 0, stream>>>(hbf0, Wb, features, b_ih, b_hh, h1, hbf1);
  k_gru_round<<<grid, dim3(256), 0, stream>>>(hbf1, Wb, h1,       b_ih, b_hh, h2, hbf0);
  k_fc<<<dim3(BS / 8), dim3(256), 0, stream>>>(h2, fc_w, fc_b, out);
}
